// EfficientRoPEAttention_80942953660878
// MI455X (gfx1250) — compile-verified, matrix operand order checked by reading, not run on hardware
//
#include <hip/hip_runtime.h>
#include <hip/hip_bf16.h>
#include <math.h>

// ---------------------------------------------------------------------------
// EfficientRoPEAttention (SAM2-style) for MI455X / gfx1250, wave32 + WMMA f16.
// All GEMMs use v_wmma_f32_16x16x32_f16 (f16 in, f32 accum). Fragments are
// loaded as two b128 transfers per lane (layout per CDNA5 ISA):
// lane l -> row (l&15), 16B segment (l>>4), segments at +0 and +32 bytes.
// Weights and V-landmarks are stored pre-transposed so every WMMA operand is
// a contiguous row-major b128 load. K/V/Q tile staging in the flash loop uses
// GLOBAL_LOAD_ASYNC_TO_LDS_B128 (ASYNCcnt) when the toolchain exposes it.
// ---------------------------------------------------------------------------

typedef _Float16 half_t;
typedef __attribute__((ext_vector_type(16))) _Float16 v16h;
typedef __attribute__((ext_vector_type(8)))  _Float16 v8h;
typedef __attribute__((ext_vector_type(8)))  float    v8f;
typedef int v4i __attribute__((vector_size(16)));   // matches builtin signature

#define B_      4
#define NQ_     4096
#define EMB_    256
#define KVIN_   64
#define FRAMES_ 7
#define NKR_    (FRAMES_ * NQ_)         // 28672 rope'd kv tokens
#define NEX_    64                      // excluded (no-rope) tail tokens
#define NK_     (NKR_ + NEX_)           // 28736
#define NKL_    (FRAMES_ * 1024 + NEX_) // 7232 pooled kv length
#define DS_     256
#define POOL_OFF 1.3862943611198906f    // 2*ln(2)

#if defined(__has_builtin)
#if __has_builtin(__builtin_amdgcn_global_load_async_to_lds_b128) && \
    __has_builtin(__builtin_amdgcn_s_wait_asynccnt)
#define HAVE_ASYNC_LDS 1
#endif
#endif

#ifdef HAVE_ASYNC_LDS
// LDS aperture: flat addr[31:0] is the LDS byte offset (ISA 10.2), so the
// AS(3) pointer is the truncated flat address; globals go int -> AS(1).
__device__ __forceinline__ void async_b128(const void* g, const void* l) {
  __builtin_amdgcn_global_load_async_to_lds_b128(
      (__attribute__((address_space(1))) v4i*)(size_t)g,
      (__attribute__((address_space(3))) v4i*)(unsigned)(size_t)l, 0, 0);
}
#endif

// Row-major 16x32 f16 fragment (A-style / B^T-style): two b128 loads per lane.
__device__ __forceinline__ v16h frag_load_rm(const half_t* __restrict__ p, int ld) {
  const int m = threadIdx.x & 15;
  const int h = (threadIdx.x >> 4) & 1;
  const half_t* row = p + m * ld + h * 8;
  v8h lo = *(const v8h*)(row);        // K = h*8 + 0..7
  v8h hi = *(const v8h*)(row + 16);   // K = 16 + h*8 + 0..7
  return __builtin_shufflevector(lo, hi, 0, 1, 2, 3, 4, 5, 6, 7,
                                 8, 9, 10, 11, 12, 13, 14, 15);
}

__device__ __forceinline__ v8f wmma_f16(v16h a, v16h b, v8f c) {
  return __builtin_amdgcn_wmma_f32_16x16x32_f16(false, a, false, b, (short)0, c,
                                                false, false);
}

__device__ __forceinline__ v8h pack8(const float* s) {
  float4 f0 = *(const float4*)(s);
  float4 f1 = *(const float4*)(s + 4);
  v8h hv;
  hv[0] = (half_t)f0.x; hv[1] = (half_t)f0.y; hv[2] = (half_t)f0.z; hv[3] = (half_t)f0.w;
  hv[4] = (half_t)f1.x; hv[5] = (half_t)f1.y; hv[6] = (half_t)f1.z; hv[7] = (half_t)f1.w;
  return hv;
}

// ---------------------------------------------------------------------------
// Prep kernels
// ---------------------------------------------------------------------------
// s: K x N row-major f32  ->  d: N x K row-major f16 (transposed weights)
__global__ void cvt_transpose_f32_f16(const float* __restrict__ s,
                                      half_t* __restrict__ d, int K, int N) {
  int i = blockIdx.x * 256 + threadIdx.x;
  if (i < K * N) {
    int kk = i / N, n = i - kk * N;
    d[n * K + kk] = (half_t)s[i];
  }
}

// axial RoPE table: t in [0,4096), pair channel c in [0,128).
__global__ void rope_table(float* __restrict__ ct, float* __restrict__ st) {
  int t = blockIdx.x;
  int c = threadIdx.x;
  float f   = __powf(10000.0f, -(float)(c & 63) / 64.0f);
  float pos = (c < 64) ? (float)(t & 63) : (float)(t >> 6);
  float a = pos * f;
  ct[t * 128 + c] = __cosf(a);
  st[t * 128 + c] = __sinf(a);
}

// ---------------------------------------------------------------------------
// Q projection (16384x256 @ 256x256) + fused RoPE, f16 out.
// 256 thr = 8 waves; 64-row tile; wave (wr,wc) owns 16 rows x 128 cols.
// ---------------------------------------------------------------------------
__global__ __launch_bounds__(256) void qproj_rope(
    const float* __restrict__ q, const half_t* __restrict__ WqT,
    const float* __restrict__ bq, const float* __restrict__ ct,
    const float* __restrict__ st, half_t* __restrict__ qr) {
  __shared__ __align__(16) half_t As[64 * 32];
  const int tid = threadIdx.x;
  const int wave = tid >> 5, wr = wave & 3, wc = wave >> 2;
  const int lane15 = tid & 15, hh = (tid >> 4) & 1;
  const int rowBase = blockIdx.x * 64;

  v8f acc[8] = {};
  for (int kc = 0; kc < 8; ++kc) {
    __syncthreads();
    {                                       // stage A 64x32: 8 f32 -> v8h/thread
      int e8 = tid * 8;
      int rr = e8 >> 5, cc = e8 & 31;
      *(v8h*)(As + e8) = pack8(q + (size_t)(rowBase + rr) * EMB_ + kc * 32 + cc);
    }
    __syncthreads();

    v16h a = frag_load_rm(As + wr * 16 * 32, 32);
#pragma unroll
    for (int nt = 0; nt < 8; ++nt) {
      v16h bf = frag_load_rm(WqT + (size_t)(wc * 128 + nt * 16) * 256 + kc * 32, 256);
      acc[nt] = wmma_f16(a, bf, acc[nt]);
    }
  }
  // epilogue: bias + RoPE via lane-pair shuffle, f16 store
#pragma unroll
  for (int nt = 0; nt < 8; ++nt) {
#pragma unroll
    for (int r = 0; r < 8; ++r) {
      int n = wc * 128 + nt * 16 + lane15;
      int row = rowBase + wr * 16 + r + 8 * hh;
      int t = row & (NQ_ - 1);
      float vb = acc[nt][r] + bq[n];
      float other = __shfl_xor(vb, 1, 32);
      int c = n >> 1;
      float cv = ct[t * 128 + c], sv = st[t * 128 + c];
      float res = (n & 1) ? (other * sv + vb * cv) : (vb * cv - other * sv);
      qr[(size_t)row * EMB_ + n] = (half_t)res;
    }
  }
}

// ---------------------------------------------------------------------------
// K/V projection (K=64) + optional RoPE + fused 2x2 landmark pooling.
// Grid (32 tiles, 7 frames, 4 batches); 128-token tile = 2 grid rows so the
// tile pools entirely in LDS. transposeOut=1 writes pooled output DS-major.
// ---------------------------------------------------------------------------
__global__ __launch_bounds__(256) void kvproj_pool(
    const float* __restrict__ in, const half_t* __restrict__ WT16,
    const float* __restrict__ bias, const float* __restrict__ ct,
    const float* __restrict__ st, half_t* __restrict__ out,
    int do_rope, float poolOff, int transposeOut) {
  __shared__ __align__(16) char smraw[65536];
  half_t* As  = (half_t*)smraw;              // 128x64  (16 KB)
  half_t* rot = (half_t*)smraw;              // 128x256 (64 KB, alias after GEMM)
  const int g = blockIdx.x, f = blockIdx.y, b = blockIdx.z;
  const int tid = threadIdx.x;
  const int wave = tid >> 5;                 // row strip (8 x 16 rows)
  const int lane15 = tid & 15, hh = (tid >> 4) & 1;
  const size_t inRow0 = (size_t)b * NK_ + (size_t)f * NQ_ + g * 128;

#pragma unroll
  for (int i = 0; i < 4; ++i) {              // stage A 128x64: 32 f32/thread
    int e8 = tid * 8 + i * 2048;
    int rr = e8 >> 6, cc = e8 & 63;
    *(v8h*)(As + e8) = pack8(in + (inRow0 + rr) * KVIN_ + cc);
  }
  __syncthreads();

  v8f acc[16] = {};
#pragma unroll
  for (int kc = 0; kc < 2; ++kc) {
    v16h a = frag_load_rm(As + (wave * 16) * 64 + kc * 32, 64);
#pragma unroll
    for (int nt = 0; nt < 16; ++nt) {
      v16h bf = frag_load_rm(WT16 + (size_t)(nt * 16) * 64 + kc * 32, 64);
      acc[nt] = wmma_f16(a, bf, acc[nt]);
    }
  }
  __syncthreads();                           // A dead; rot aliases it

  const int tBase = g * 128;                 // frame-local token base
#pragma unroll
  for (int nt = 0; nt < 16; ++nt) {
#pragma unroll
    for (int r = 0; r < 8; ++r) {
      int n = nt * 16 + lane15;
      int lr = wave * 16 + r + 8 * hh;       // local row 0..127
      float vb = acc[nt][r] + bias[n];
      float other = __shfl_xor(vb, 1, 32);   // uniform exec, always run
      float res = vb;
      if (do_rope) {
        int t = tBase + lr;
        int c = n >> 1;
        float cv = ct[t * 128 + c], sv = st[t * 128 + c];
        res = (n & 1) ? (other * sv + vb * cv) : (vb * cv - other * sv);
      }
      rot[lr * 256 + n] = (half_t)res;
    }
  }
  __syncthreads();

  // 2x2 pool: tokens {2X, 2X+1, 64+2X, 65+2X} -> pooled token f*1024+g*32+X
  const int pTok0 = f * 1024 + g * 32;
  if (!transposeOut) {
#pragma unroll
    for (int i = 0; i < 4; ++i) {
      int o8 = tid * 8 + i * 2048;           // 8192 outputs, 8/thread/iter
      int X = o8 >> 8, col0 = o8 & 255;
      v8h a0 = *(const v8h*)(rot + (2 * X) * 256 + col0);
      v8h a1 = *(const v8h*)(rot + (2 * X + 1) * 256 + col0);
      v8h a2 = *(const v8h*)(rot + (64 + 2 * X) * 256 + col0);
      v8h a3 = *(const v8h*)(rot + (65 + 2 * X) * 256 + col0);
      v8h hv;
#pragma unroll
      for (int j = 0; j < 8; ++j)
        hv[j] = (half_t)(((float)a0[j] + (float)a1[j] + (float)a2[j] + (float)a3[j]) *
                             0.25f + poolOff);
      *(v8h*)(out + ((size_t)b * NKL_ + pTok0 + X) * DS_ + col0) = hv;
    }
  } else {
#pragma unroll
    for (int i = 0; i < 32; ++i) {
      int e = tid + i * 256;
      int X = e >> 8, col = e & 255;
      float s4 = (float)rot[(2 * X) * 256 + col] + (float)rot[(2 * X + 1) * 256 + col] +
                 (float)rot[(64 + 2 * X) * 256 + col] + (float)rot[(65 + 2 * X) * 256 + col];
      out[((size_t)b * DS_ + col) * NKL_ + pTok0 + X] = (half_t)(s4 * 0.25f + poolOff);
    }
  }
}

// no-rope, no-pool tail (64 tokens/batch): tiny, scalar f32 dot products.
__global__ void tail_proj(const float* __restrict__ in, const float* __restrict__ W,
                          const float* __restrict__ bias, half_t* __restrict__ out,
                          int transposeOut) {
  int idx = blockIdx.x * 256 + threadIdx.x;  // B*64*256 = 65536
  int n = idx & 255, r = (idx >> 8) & 63, b = idx >> 14;
  float s = bias[n];
  const float* row = in + ((size_t)b * NK_ + NKR_ + r) * KVIN_;
#pragma unroll
  for (int kk = 0; kk < KVIN_; ++kk) s += row[kk] * W[kk * EMB_ + n];
  if (transposeOut)
    out[((size_t)b * DS_ + n) * NKL_ + FRAMES_ * 1024 + r] = (half_t)s;
  else
    out[((size_t)b * NKL_ + FRAMES_ * 1024 + r) * DS_ + n] = (half_t)s;
}

// ---------------------------------------------------------------------------
// Flash attention over pooled KV (len 7232 = 226 chunks of 32).
// 128 thr = 4 waves; each wave owns 16 Q rows x full DS=256 accumulator.
// K is token-major (QK^T B-frags = row-major K rows); V is DS-major so the
// P.V B-frags are contiguous b128 LDS loads from the transposed chunk.
// ---------------------------------------------------------------------------
__global__ __launch_bounds__(128) void attn_kernel(
    const half_t* __restrict__ qr, const half_t* __restrict__ klm,
    const half_t* __restrict__ vlmT, half_t* __restrict__ ao) {
  __shared__ __align__(16) char sm[65536];
  half_t* Qs  = (half_t*)sm;                 // 64x256  (32 KB)
  half_t* Ks  = (half_t*)(sm + 32768);       // 32x256  (16 KB)
  half_t* VsT = (half_t*)(sm + 49152);       // 256x32  (16 KB), DS-major
  const int b = blockIdx.y;
  const int qbase = blockIdx.x * 64;
  const int tid = threadIdx.x;
  const int wave = tid >> 5;
  const int lane15 = tid & 15, hh = (tid >> 4) & 1;
  half_t* Pw = Ks + wave * 512;              // 16x32 P patch, aliases spent K

  {                                          // stage Q tile once (async if avail)
    const half_t* src = qr + ((size_t)b * NQ_ + qbase) * DS_;
#ifdef HAVE_ASYNC_LDS
#pragma unroll
    for (int i = 0; i < 16; ++i)
      async_b128(src + tid * 8 + i * 1024, Qs + tid * 8 + i * 1024);
    __builtin_amdgcn_s_wait_asynccnt(0);
#else
#pragma unroll
    for (int i = 0; i < 16; ++i)
      ((v8h*)Qs)[tid + i * 128] = ((const v8h*)src)[tid + i * 128];
#endif
  }

  v8f o[16] = {};
  float m_i[8], l_i[8];
#pragma unroll
  for (int r = 0; r < 8; ++r) { m_i[r] = -1e30f; l_i[r] = 0.0f; }
  const float scale = 0.0625f;               // 1/sqrt(256)

  const half_t* kbase = klm + (size_t)b * NKL_ * DS_;
  const half_t* vbase = vlmT + (size_t)b * DS_ * NKL_;

  for (int ck = 0; ck < NKL_ / 32; ++ck) {
    __syncthreads();                         // prior P/V reads done
    {
      const half_t* ks = kbase + (size_t)ck * 32 * DS_;
#ifdef HAVE_ASYNC_LDS
#pragma unroll
      for (int i = 0; i < 8; ++i)
        async_b128(ks + tid * 8 + i * 1024, Ks + tid * 8 + i * 1024);
#pragma unroll
      for (int i = 0; i < 8; ++i) {          // V transposed: row n, 32 tokens
        int e = tid + i * 128;
        int n = e >> 2, seg = e & 3;
        async_b128(vbase + (size_t)n * NKL_ + ck * 32 + seg * 8, VsT + e * 8);
      }
      __builtin_amdgcn_s_wait_asynccnt(0);
#else
#pragma unroll
      for (int i = 0; i < 8; ++i)
        ((v8h*)Ks)[tid + i * 128] = ((const v8h*)ks)[tid + i * 128];
#pragma unroll
      for (int i = 0; i < 8; ++i) {
        int e = tid + i * 128;
        int n = e >> 2, seg = e & 3;
        ((v8h*)VsT)[e] = *(const v8h*)(vbase + (size_t)n * NKL_ + ck * 32 + seg * 8);
      }
#endif
    }
    // prefetch next chunk while this one is consumed (global_prefetch_b8)
    if (ck + 1 < NKL_ / 32) {
      __builtin_prefetch((const char*)(kbase + (size_t)(ck + 1) * 32 * DS_) + tid * 128, 0, 1);
      __builtin_prefetch(vbase + (size_t)(tid * 2) * NKL_ + (ck + 1) * 32, 0, 1);
    }
    __syncthreads();

    // S = Q . K^T  (two 16x16 tiles over K=256)
    v8f s0 = {}, s1 = {};
#pragma unroll
    for (int kc = 0; kc < 8; ++kc) {
      v16h aq = frag_load_rm(Qs + wave * 16 * 256 + kc * 32, 256);
      v16h b0 = frag_load_rm(Ks + 0 * 256 + kc * 32, 256);
      v16h b1 = frag_load_rm(Ks + 16 * 256 + kc * 32, 256);
      s0 = wmma_f16(aq, b0, s0);
      s1 = wmma_f16(aq, b1, s1);
    }
    __syncthreads();                         // K reads done before P aliases it

    // online softmax per row (rows live in vgpr index, cols in lanes)
#pragma unroll
    for (int r = 0; r < 8; ++r) {
      float a0 = s0[r] * scale, a1 = s1[r] * scale;
      float mx = fmaxf(a0, a1);
      mx = fmaxf(mx, __shfl_xor(mx, 1, 32));
      mx = fmaxf(mx, __shfl_xor(mx, 2, 32));
      mx = fmaxf(mx, __shfl_xor(mx, 4, 32));
      mx = fmaxf(mx, __shfl_xor(mx, 8, 32));
      float mnew = fmaxf(m_i[r], mx);
      float alpha = __expf(m_i[r] - mnew);
      float p0 = __expf(a0 - mnew), p1 = __expf(a1 - mnew);
      float rs = p0 + p1;
      rs += __shfl_xor(rs, 1, 32);
      rs += __shfl_xor(rs, 2, 32);
      rs += __shfl_xor(rs, 4, 32);
      rs += __shfl_xor(rs, 8, 32);
      l_i[r] = l_i[r] * alpha + rs;
      m_i[r] = mnew;
#pragma unroll
      for (int nt = 0; nt < 16; ++nt) o[nt][r] *= alpha;
      int rr = r + 8 * hh;
      Pw[rr * 32 + lane15]      = (half_t)p0;
      Pw[rr * 32 + 16 + lane15] = (half_t)p1;
    }

    // O += P . V   (B-frags: contiguous rows of the DS-major V chunk)
    v16h pa = frag_load_rm(Pw, 32);
#pragma unroll
    for (int nt = 0; nt < 16; ++nt) {
      v16h vb = frag_load_rm(VsT + (nt * 16) * 32, 32);
      o[nt] = wmma_f16(pa, vb, o[nt]);
    }
  }

  half_t* dst = ao + ((size_t)b * NQ_ + qbase + wave * 16) * DS_;
#pragma unroll
  for (int nt = 0; nt < 16; ++nt) {
#pragma unroll
    for (int r = 0; r < 8; ++r) {
      int rr = r + 8 * hh;
      dst[rr * DS_ + nt * 16 + lane15] = (half_t)(o[nt][r] / l_i[r]);
    }
  }
}

// ---------------------------------------------------------------------------
// Output projection (f16 in, f32 out), same tiling as qproj.
// ---------------------------------------------------------------------------
__global__ __launch_bounds__(256) void oproj(
    const half_t* __restrict__ ao, const half_t* __restrict__ WoT,
    const float* __restrict__ bo, float* __restrict__ outp) {
  __shared__ __align__(16) half_t As[64 * 32];
  const int tid = threadIdx.x;
  const int wave = tid >> 5, wr = wave & 3, wc = wave >> 2;
  const int lane15 = tid & 15, hh = (tid >> 4) & 1;
  const int rowBase = blockIdx.x * 64;

  v8f acc[8] = {};
  for (int kc = 0; kc < 8; ++kc) {
    __syncthreads();
    {                                        // A already f16: one v8h copy/thread
      int e8 = tid * 8;
      int rr = e8 >> 5, cc = e8 & 31;
      *(v8h*)(As + e8) =
          *(const v8h*)(ao + (size_t)(rowBase + rr) * EMB_ + kc * 32 + cc);
    }
    __syncthreads();

    v16h a = frag_load_rm(As + wr * 16 * 32, 32);
#pragma unroll
    for (int nt = 0; nt < 8; ++nt) {
      v16h bf = frag_load_rm(WoT + (size_t)(wc * 128 + nt * 16) * 256 + kc * 32, 256);
      acc[nt] = wmma_f16(a, bf, acc[nt]);
    }
  }
#pragma unroll
  for (int nt = 0; nt < 8; ++nt) {
#pragma unroll
    for (int r = 0; r < 8; ++r) {
      int n = wc * 128 + nt * 16 + lane15;
      int row = rowBase + wr * 16 + r + 8 * hh;
      outp[(size_t)row * EMB_ + n] = acc[nt][r] + bo[n];
    }
  }
}

// ---------------------------------------------------------------------------
extern "C" void kernel_launch(void* const* d_in, const int* in_sizes, int n_in,
                              void* d_out, int out_size, void* d_ws, size_t ws_size,
                              hipStream_t stream) {
  (void)in_sizes; (void)n_in; (void)out_size; (void)ws_size;
  const float* q  = (const float*)d_in[0];
  const float* k  = (const float*)d_in[1];
  const float* v  = (const float*)d_in[2];
  const float* Wq = (const float*)d_in[3];
  const float* bq = (const float*)d_in[4];
  const float* Wk = (const float*)d_in[5];
  const float* bk = (const float*)d_in[6];
  const float* Wv = (const float*)d_in[7];
  const float* bv = (const float*)d_in[8];
  const float* Wo = (const float*)d_in[9];
  const float* bo = (const float*)d_in[10];

  char* ws = (char*)d_ws;
  size_t off = 0;
  half_t* WqT  = (half_t*)(ws + off); off += 256 * 256 * 2;          // N x K
  half_t* WkT  = (half_t*)(ws + off); off += 64 * 256 * 2;           // 256 x 64
  half_t* WvT  = (half_t*)(ws + off); off += 64 * 256 * 2;
  half_t* WoT  = (half_t*)(ws + off); off += 256 * 256 * 2;
  float*  ct   = (float*)(ws + off);  off += (size_t)NQ_ * 128 * 4;  // 2 MB
  float*  st   = (float*)(ws + off);  off += (size_t)NQ_ * 128 * 4;  // 2 MB
  half_t* qr   = (half_t*)(ws + off); off += (size_t)B_ * NQ_ * DS_ * 2;
  half_t* klm  = (half_t*)(ws + off); off += (size_t)B_ * NKL_ * DS_ * 2; // token-major
  half_t* vlmT = (half_t*)(ws + off); off += (size_t)B_ * DS_ * NKL_ * 2; // DS-major
  half_t* ao   = (half_t*)(ws + off); off += (size_t)B_ * NQ_ * DS_ * 2;

  cvt_transpose_f32_f16<<<(256 * 256 + 255) / 256, 256, 0, stream>>>(Wq, WqT, 256, 256);
  cvt_transpose_f32_f16<<<(64 * 256 + 255) / 256, 256, 0, stream>>>(Wk, WkT, 64, 256);
  cvt_transpose_f32_f16<<<(64 * 256 + 255) / 256, 256, 0, stream>>>(Wv, WvT, 64, 256);
  cvt_transpose_f32_f16<<<(256 * 256 + 255) / 256, 256, 0, stream>>>(Wo, WoT, 256, 256);
  rope_table<<<NQ_, 128, 0, stream>>>(ct, st);

  qproj_rope<<<(B_ * NQ_) / 64, 256, 0, stream>>>(q, WqT, bq, ct, st, qr);

  kvproj_pool<<<dim3(32, FRAMES_, B_), 256, 0, stream>>>(k, WkT, bk, ct, st,
                                                         klm, 1, POOL_OFF, 0);
  kvproj_pool<<<dim3(32, FRAMES_, B_), 256, 0, stream>>>(v, WvT, bv, ct, st,
                                                         vlmT, 0, 0.0f, 1);
  tail_proj<<<(B_ * NEX_ * EMB_) / 256, 256, 0, stream>>>(k, Wk, bk, klm, 0);
  tail_proj<<<(B_ * NEX_ * EMB_) / 256, 256, 0, stream>>>(v, Wv, bv, vlmT, 1);

  attn_kernel<<<dim3(NQ_ / 64, B_), 128, 0, stream>>>(qr, klm, vlmT, ao);

  oproj<<<(B_ * NQ_) / 64, 256, 0, stream>>>(ao, WoT, bo, (float*)d_out);
}
